// RTLayer_5652176962163
// MI455X (gfx1250) — matrix operand order repaired and checked by reading, not compiled, not run on hardware
//
#include <hip/hip_runtime.h>
#include <math.h>

typedef __attribute__((ext_vector_type(16))) __bf16 v16bf;
typedef __attribute__((ext_vector_type(4)))  __bf16 v4bf;
typedef __attribute__((ext_vector_type(8)))  float  v8f;

#define B_  2
#define N_  384
#define D_  128
#define H_  8

// ---------- helpers ----------
__device__ inline float gelu_exact(float x) {
  return 0.5f * x * (1.0f + erff(x * 0.7071067811865476f));
}
__device__ inline v8f wmma_bf16(v16bf a, v16bf b, v8f c) {
  return __builtin_amdgcn_wmma_f32_16x16x32_bf16(false, a, false, b, (short)0, c, false, false);
}
// DPP-based add over 16-lane row (no DS traffic): xor1, xor2, then quad rotations
template <int CTRL>
__device__ inline float dpp_addf(float x) {
  int y = __builtin_amdgcn_update_dpp(0, __float_as_int(x), CTRL, 0xF, 0xF, true);
  return x + __int_as_float(y);
}
__device__ inline float row16_sum(float p) {
  p = dpp_addf<0xB1>(p);    // quad_perm [1,0,3,2]  : lane^1
  p = dpp_addf<0x4E>(p);    // quad_perm [2,3,0,1]  : lane^2
  p = dpp_addf<0x124>(p);   // row_ror:4            : +quad
  p = dpp_addf<0x128>(p);   // row_ror:8            : +2 quads
  return p;                 // all 16 lanes of the row hold the total
}
// B fragment: packed layout [ct][ks][lane][16e], 32B-aligned -> two b128 loads
__device__ inline v16bf load_bfrag(const __bf16* pk, int ct, int ks, int KS, int lane) {
  return *reinterpret_cast<const v16bf*>(pk + (((size_t)(ct * KS + ks) * 32 + lane) << 4));
}
// A fragment layout (16-bit A 16x32 per ISA 7.12.2):
// lane L: M = L&15 ; elem e: K = ks*32 + (e>>3)*16 + ((L>>4)<<3) + (e&7)
__device__ inline int fragidx(int m, int k) {
  int s  = k >> 5;
  int kk = k & 31;
  int g  = kk >> 4;
  int hi = (kk >> 3) & 1;
  int w  = kk & 7;
  int e  = (g << 3) + w;
  int lane = (hi << 4) + m;
  return ((s << 5) + lane) * 16 + e;
}
__device__ inline v16bf load_afrag(const __bf16* frag, int ks, int lane) {
  return *reinterpret_cast<const v16bf*>(frag + (((ks << 5) + lane) << 4));
}
// pack 4 consecutive-K bf16 (aligned 4, never straddles a sub-block) -> one 8B store
__device__ inline void store_bf4(__bf16* dst, float4 v) {
  v4bf p;
  p[0] = (__bf16)v.x; p[1] = (__bf16)v.y; p[2] = (__bf16)v.z; p[3] = (__bf16)v.w;
  *reinterpret_cast<v4bf*>(dst) = p;
}

// ---------- kernel 0: pack weights (K x NC, row-major f32) to bf16 fragment order ----------
__global__ void pack_w(const float* __restrict__ W, __bf16* __restrict__ dst,
                       int K, int NC) {
  int KS = K >> 5;
  int total = K * NC;
  for (int p = blockIdx.x * blockDim.x + threadIdx.x; p < total; p += gridDim.x * blockDim.x) {
    int e = p & 15;
    int lane = (p >> 4) & 31;
    int rest = p >> 9;
    int ks = rest % KS;
    int ct = rest / KS;
    int k = ks * 32 + ((e >> 3) << 4) + ((lane >> 4) << 3) + (e & 7);
    int n = (ct << 4) + (lane & 15);
    dst[p] = (__bf16)W[k * NC + n];
  }
}

// ---------- kernel 1: qkv_n = node @ wqkv_n ----------
__global__ void node_qkv(const float* __restrict__ node, const float* __restrict__ wn,
                         float* __restrict__ qkvn) {
  int idx = blockIdx.x * blockDim.x + threadIdx.x;
  if (idx >= B_ * N_ * 3 * D_) return;
  int c = idx % (3 * D_);
  int bn = idx / (3 * D_);
  const float* row = node + (size_t)bn * D_;
  float s = 0.f;
  for (int k = 0; k < D_; ++k) s += row[k] * wn[k * (3 * D_) + c];
  qkvn[idx] = s;
}

// ---------- kernel 1b: rearrange kn/vn to [b][h][dl][j] for contiguous per-lane loads ----------
__global__ void qkv_rearr(const float* __restrict__ qkvn,
                          float* __restrict__ kn2, float* __restrict__ vn2) {
  int idx = blockIdx.x * blockDim.x + threadIdx.x;   // B*H*16*384
  if (idx >= B_ * H_ * 16 * N_) return;
  int j = idx % N_;
  int rest = idx / N_;
  int dl = rest % 16;
  int bh = rest / 16;
  int h = bh % H_, b = bh / H_;
  const float* src = qkvn + ((size_t)(b * N_ + j)) * (3 * D_) + h * 48;
  kn2[idx] = src[16 + dl];
  vn2[idx] = src[32 + dl];
}

// ---------- kernel 2: fused RT attention, one block per (b,i), wave w = head w ----------
__global__ void __launch_bounds__(256) attn_fused(const float* __restrict__ qkvn,
                                                  const float* __restrict__ kn2,
                                                  const float* __restrict__ vn2,
                                                  const float* __restrict__ edge,
                                                  const __bf16* __restrict__ wpk,
                                                  float* __restrict__ attn_out) {
  __shared__ float dots[H_][N_];                 // 12 KB
  __shared__ __bf16 fragE[12 * 4 * 32 * 16];     // 48 KB: 12 j-tiles, frag order
  int b = blockIdx.x / N_, i = blockIdx.x % N_;
  int tid = threadIdx.x;
  int lane = tid & 31;
  int h = tid >> 5;
  int dl = lane & 15, hi = lane >> 4;
  const float scale = 0.08838834764831845f;      // 1/sqrt(128)

  float qn_i = qkvn[((size_t)(b * N_ + i)) * (3 * D_) + h * 48 + dl];
  const float* ebase = edge + ((size_t)(b * N_ + i)) * N_ * D_;
  const float* knrow = kn2 + ((size_t)((b * H_ + h) * 16 + dl)) * N_;
  const float* vnrow = vn2 + ((size_t)((b * H_ + h) * 16 + dl)) * N_;

  v16bf bq[4], bk[4];
#pragma unroll
  for (int s = 0; s < 4; ++s) {
    bq[s] = load_bfrag(wpk, 4 * h + 0, s, 4, lane);
    bk[s] = load_bfrag(wpk, 4 * h + 1, s, 4, lane);
  }

  // ---- pass 1: dots = scale * (qn+eq)·(kn+ek) ----
  for (int half = 0; half < 2; ++half) {
    // cooperative float4 stage of 192 j-rows into fragment order
    for (int t4 = tid; t4 < 192 * 32; t4 += 256) {
      int jloc = t4 >> 5, k = (t4 & 31) << 2;
      float4 v = *reinterpret_cast<const float4*>(
          ebase + (size_t)(half * 192 + jloc) * D_ + k);
      int jt_loc = jloc >> 4, m = jloc & 15;
      store_bf4(fragE + jt_loc * (4 * 32 * 16) + fragidx(m, k), v);
    }
    __syncthreads();
    for (int jl = 0; jl < 12; ++jl) {
      int jt = half * 12 + jl;
      const __bf16* fr = fragE + jl * (4 * 32 * 16);
      v8f aq = {}, ak = {};
#pragma unroll
      for (int s = 0; s < 4; ++s) {
        v16bf a = load_afrag(fr, s, lane);
        aq = wmma_bf16(a, bq[s], aq);
        ak = wmma_bf16(a, bk[s], ak);
      }
      const float4* kv = reinterpret_cast<const float4*>(knrow + jt * 16 + (hi << 3));
      float4 k0 = kv[0], k1 = kv[1];
      float kn8[8] = {k0.x, k0.y, k0.z, k0.w, k1.x, k1.y, k1.z, k1.w};
#pragma unroll
      for (int r = 0; r < 8; ++r) {
        float p = (qn_i + aq[r]) * (kn8[r] + ak[r]);
        p = row16_sum(p);
        if (dl == r) dots[h][jt * 16 + r + (hi << 3)] = p * scale;
      }
    }
    __syncthreads();
  }

  // ---- softmax over j (per-wave, own head) ----
  float mx = -3.4e38f;
  for (int t = lane; t < N_; t += 32) mx = fmaxf(mx, dots[h][t]);
#pragma unroll
  for (int m = 1; m < 32; m <<= 1) mx = fmaxf(mx, __shfl_xor(mx, m, 32));
  float sum = 0.f;
  for (int t = lane; t < N_; t += 32) {
    float ev = expf(dots[h][t] - mx);
    dots[h][t] = ev;
    sum += ev;
  }
#pragma unroll
  for (int m = 1; m < 32; m <<= 1) sum += __shfl_xor(sum, m, 32);
  float inv = 1.0f / sum;
  for (int t = lane; t < N_; t += 32) dots[h][t] *= inv;
  __syncthreads();

  // ---- pass 2: out = sum_j attn * (vn*em + ev) ----
  v16bf bv[4], bm[4];
#pragma unroll
  for (int s = 0; s < 4; ++s) {
    bv[s] = load_bfrag(wpk, 4 * h + 2, s, 4, lane);
    bm[s] = load_bfrag(wpk, 4 * h + 3, s, 4, lane);
  }
  float oacc = 0.f;
  for (int half = 0; half < 2; ++half) {
    for (int t4 = tid; t4 < 192 * 32; t4 += 256) {
      int jloc = t4 >> 5, k = (t4 & 31) << 2;
      float4 v = *reinterpret_cast<const float4*>(
          ebase + (size_t)(half * 192 + jloc) * D_ + k);
      int jt_loc = jloc >> 4, m = jloc & 15;
      store_bf4(fragE + jt_loc * (4 * 32 * 16) + fragidx(m, k), v);
    }
    __syncthreads();
    for (int jl = 0; jl < 12; ++jl) {
      int jt = half * 12 + jl;
      const __bf16* fr = fragE + jl * (4 * 32 * 16);
      v8f av = {}, am = {};
#pragma unroll
      for (int s = 0; s < 4; ++s) {
        v16bf a = load_afrag(fr, s, lane);
        av = wmma_bf16(a, bv[s], av);
        am = wmma_bf16(a, bm[s], am);
      }
      const float4* vv = reinterpret_cast<const float4*>(vnrow + jt * 16 + (hi << 3));
      float4 v0 = vv[0], v1 = vv[1];
      float vn8[8] = {v0.x, v0.y, v0.z, v0.w, v1.x, v1.y, v1.z, v1.w};
      const float4* dd = reinterpret_cast<const float4*>(&dots[h][jt * 16 + (hi << 3)]);
      float4 d0 = dd[0], d1 = dd[1];
      float at8[8] = {d0.x, d0.y, d0.z, d0.w, d1.x, d1.y, d1.z, d1.w};
#pragma unroll
      for (int r = 0; r < 8; ++r)
        oacc += at8[r] * (vn8[r] * am[r] + av[r]);
    }
    __syncthreads();
  }
  oacc += __shfl_xor(oacc, 16, 32);
  if (lane < 16) attn_out[((size_t)(b * N_ + i)) * D_ + h * 16 + lane] = oacc;
}

// ---------- kernel 3: node update ----------
__global__ void __launch_bounds__(128) node_update(
    const float* __restrict__ node, const float* __restrict__ attn_o,
    const float* __restrict__ lw, const float* __restrict__ lb,
    const float* __restrict__ g0, const float* __restrict__ bb0,
    const float* __restrict__ g1, const float* __restrict__ bb1,
    const float* __restrict__ w1, const float* __restrict__ w2, const float* __restrict__ b2,
    float* __restrict__ out_x) {
  __shared__ float sa[128], sx[128], sh2[256], red[128];
  int bn = blockIdx.x, t = threadIdx.x;
  sa[t] = attn_o[(size_t)bn * 128 + t];
  __syncthreads();
  float acc = lb[t];
  for (int k = 0; k < 128; ++k) acc += sa[k] * lw[k * 128 + t];
  float r1 = node[(size_t)bn * 128 + t] + acc;
  red[t] = r1; __syncthreads();
  for (int s = 64; s; s >>= 1) { if (t < s) red[t] += red[t + s]; __syncthreads(); }
  float mean = red[0] * (1.f / 128.f); __syncthreads();
  float dv = r1 - mean;
  red[t] = dv * dv; __syncthreads();
  for (int s = 64; s; s >>= 1) { if (t < s) red[t] += red[t + s]; __syncthreads(); }
  float rstd = rsqrtf(red[0] * (1.f / 128.f) + 1e-5f); __syncthreads();
  float x = dv * rstd * g0[t] + bb0[t];
  sx[t] = x; __syncthreads();
  float h0 = 0.f, h1 = 0.f;
  for (int k = 0; k < 128; ++k) {
    float xv = sx[k];
    h0 += xv * w1[k * 256 + t];
    h1 += xv * w1[k * 256 + 128 + t];
  }
  sh2[t] = gelu_exact(h0);
  sh2[128 + t] = gelu_exact(h1);
  __syncthreads();
  float y2 = b2[t];
  for (int k = 0; k < 256; ++k) y2 += sh2[k] * w2[k * 128 + t];
  float r2 = x + y2;
  red[t] = r2; __syncthreads();
  for (int s = 64; s; s >>= 1) { if (t < s) red[t] += red[t + s]; __syncthreads(); }
  float mean2 = red[0] * (1.f / 128.f); __syncthreads();
  float dv2 = r2 - mean2;
  red[t] = dv2 * dv2; __syncthreads();
  for (int s = 64; s; s >>= 1) { if (t < s) red[t] += red[t + s]; __syncthreads(); }
  float rstd2 = rsqrtf(red[0] * (1.f / 128.f) + 1e-5f);
  out_x[(size_t)bn * 128 + t] = dv2 * rstd2 * g1[t] + bb1[t];
}

// ---------- kernel 4a: edge means ----------
__global__ void __launch_bounds__(128) edge_means(const float* __restrict__ edge,
                                                  float* __restrict__ mi,
                                                  float* __restrict__ mj) {
  int blk = blockIdx.x;
  int which = blk / (B_ * N_);
  int bn = blk % (B_ * N_);
  int b = bn / N_, n = bn % N_;
  int d = threadIdx.x;
  float s = 0.f;
  if (which == 0) {
    for (int i2 = 0; i2 < N_; ++i2)
      s += edge[(((size_t)(b * N_ + i2)) * N_ + n) * D_ + d];
    mi[(size_t)bn * D_ + d] = s * (1.f / (float)N_);
  } else {
    const float* rowb = edge + ((size_t)(b * N_ + n)) * N_ * D_;
    for (int j2 = 0; j2 < N_; ++j2) s += rowb[(size_t)j2 * D_ + d];
    mj[(size_t)bn * D_ + d] = s * (1.f / (float)N_);
  }
}

// ---------- kernel 4b: per-row terms sn/tn/se/te ----------
__global__ void __launch_bounds__(128) edge_terms(
    const float* __restrict__ x, const float* __restrict__ mi, const float* __restrict__ mj,
    const float* __restrict__ wS, const float* __restrict__ bS,
    const float* __restrict__ wT, const float* __restrict__ bT,
    const float* __restrict__ wec, const float* __restrict__ wer,
    float* __restrict__ sn, float* __restrict__ tn,
    float* __restrict__ se, float* __restrict__ te) {
  __shared__ float rx[128], rmi[128], rmj[128];
  int bn = blockIdx.x, t = threadIdx.x;
  rx[t]  = x[(size_t)bn * 128 + t];
  rmi[t] = mi[(size_t)bn * 128 + t];
  rmj[t] = mj[(size_t)bn * 128 + t];
  __syncthreads();
  float aS = bS[t], aT = bT[t], aSe = 0.f, aTe = 0.f;
  for (int k = 0; k < 128; ++k) {
    aS  += rx[k]  * wS[k * 128 + t];
    aT  += rx[k]  * wT[k * 128 + t];
    aSe += rmi[k] * wec[k * 128 + t];
    aTe += rmj[k] * wer[k * 128 + t];
  }
  sn[(size_t)bn * 128 + t] = aS;
  tn[(size_t)bn * 128 + t] = aT;
  se[(size_t)bn * 128 + t] = aSe;
  te[(size_t)bn * 128 + t] = aTe;
}

// ---------- kernel 5: fused edge update, one wave per 16-row j-tile of (b,i) ----------
__global__ void __launch_bounds__(32) edge_fused(
    const float* __restrict__ edge,
    const float* __restrict__ sn, const float* __restrict__ tn,
    const float* __restrict__ se, const float* __restrict__ te,
    const __bf16* __restrict__ pk_we, const float* __restrict__ be,
    const __bf16* __restrict__ pk_w1, const float* __restrict__ b1,
    const __bf16* __restrict__ pk_e1w1, const __bf16* __restrict__ pk_e1w2,
    const float* __restrict__ b2,
    const float* __restrict__ g0, const float* __restrict__ bb0,
    const float* __restrict__ g1, const float* __restrict__ bb1,
    float* __restrict__ out_e) {
  __shared__ __bf16 fragIn[8 * 32 * 16];   // concat tile (K=256), 8 KB
  __shared__ float  bufR[16 * 128];        // edge_ij residual / final t2, 8 KB
  __shared__ __bf16 fragH[4 * 32 * 16];    // gelu1 (K=128), 4 KB
  __shared__ float  bufT[16 * 128];        // t / x2 f32, 8 KB
  __shared__ __bf16 fragX2[4 * 32 * 16];   // x2 bf16, 4 KB
  __shared__ __bf16 fragH2[8 * 32 * 16];   // gelu2 (K=256), 8 KB
  __shared__ float  bufTS[16 * 128];       // tn[j]+se[j] per tile row, 8 KB
  __shared__ float  sg[4 * 128];           // eln0_g, eln0_b, eln1_g, eln1_b

  int blk = blockIdx.x;
  int jt = blk % (N_ / 16);
  int rem = blk / (N_ / 16);
  int i = rem % N_;
  int b = rem / N_;
  int lane = threadIdx.x;
  int dl = lane & 15, hi = lane >> 4;

  // stage concat(edge_ij, edge_ji) tile (float4) into fragment order + residual copy
  for (int t4 = lane; t4 < 16 * 64; t4 += 32) {
    int row = t4 >> 6, col = (t4 & 63) << 2;
    int j = jt * 16 + row;
    float4 v;
    if (col < 128) {
      v = *reinterpret_cast<const float4*>(
          edge + (((size_t)(b * N_ + i)) * N_ + j) * D_ + col);
      *reinterpret_cast<float4*>(&bufR[row * 128 + col]) = v;
    } else {
      v = *reinterpret_cast<const float4*>(
          edge + (((size_t)(b * N_ + j)) * N_ + i) * D_ + (col - 128));
    }
    store_bf4(fragIn + fragidx(row, col), v);
  }
  // stage tn[j]+se[j] rows (float4)
  for (int t4 = lane; t4 < 16 * 32; t4 += 32) {
    int row = t4 >> 5, n = (t4 & 31) << 2;
    int jb = (b * N_ + jt * 16 + row) * D_ + n;
    float4 a = *reinterpret_cast<const float4*>(tn + jb);
    float4 c = *reinterpret_cast<const float4*>(se + jb);
    float4 o; o.x = a.x + c.x; o.y = a.y + c.y; o.z = a.z + c.z; o.w = a.w + c.w;
    *reinterpret_cast<float4*>(&bufTS[row * 128 + n]) = o;
  }
  // stage LN gamma/beta
  for (int t = lane; t < 4 * 128; t += 32) {
    int which = t >> 7, n = t & 127;
    const float* src = (which == 0) ? g0 : (which == 1) ? bb0 : (which == 2) ? g1 : bb1;
    sg[t] = src[n];
  }
  __syncthreads();

  int ibase = (b * N_ + i) * D_;

  // GEMM1: concat(16x256) @ We(256x128) + biases -> gelu -> fragH
  for (int c = 0; c < 8; ++c) {
    v8f acc = {};
#pragma unroll
    for (int s = 0; s < 8; ++s)
      acc = wmma_bf16(load_afrag(fragIn, s, lane), load_bfrag(pk_we, c, s, 8, lane), acc);
    int n = c * 16 + dl;
    float cb = be[n] + sn[ibase + n] + te[ibase + n];
#pragma unroll
    for (int r = 0; r < 8; ++r) {
      int row = r + (hi << 3);
      float u = gelu_exact(acc[r] + cb + bufTS[row * 128 + n]);
      fragH[fragidx(row, n)] = (__bf16)u;
    }
  }
  __syncthreads();

  // GEMM2: gelu1(16x128) @ W1(128x128) + b1 + residual(edge_ij) -> bufT
  for (int c = 0; c < 8; ++c) {
    v8f acc = {};
#pragma unroll
    for (int s = 0; s < 4; ++s)
      acc = wmma_bf16(load_afrag(fragH, s, lane), load_bfrag(pk_w1, c, s, 4, lane), acc);
    int n = c * 16 + dl;
#pragma unroll
    for (int r = 0; r < 8; ++r) {
      int row = r + (hi << 3);
      bufT[row * 128 + n] = bufR[row * 128 + n] + acc[r] + b1[n];
    }
  }
  __syncthreads();

  // LN0 per row -> bufT (f32) and fragX2 (bf16)
  {
    int row = dl, half = hi;
    float s = 0.f, ss = 0.f;
    for (int c4 = half * 16; c4 < half * 16 + 16; ++c4) {
      float4 v = *reinterpret_cast<const float4*>(&bufT[row * 128 + c4 * 4]);
      s += v.x + v.y + v.z + v.w;
      ss += v.x * v.x + v.y * v.y + v.z * v.z + v.w * v.w;
    }
    s += __shfl_xor(s, 16, 32);
    ss += __shfl_xor(ss, 16, 32);
    float mean = s * (1.f / 128.f);
    float rstd = rsqrtf(ss * (1.f / 128.f) - mean * mean + 1e-5f);
    for (int c4 = half * 16; c4 < half * 16 + 16; ++c4) {
      int c = c4 * 4;
      float4 v = *reinterpret_cast<const float4*>(&bufT[row * 128 + c]);
      float4 gg = *reinterpret_cast<const float4*>(&sg[c]);
      float4 bb = *reinterpret_cast<const float4*>(&sg[128 + c]);
      float4 o;
      o.x = (v.x - mean) * rstd * gg.x + bb.x;
      o.y = (v.y - mean) * rstd * gg.y + bb.y;
      o.z = (v.z - mean) * rstd * gg.z + bb.z;
      o.w = (v.w - mean) * rstd * gg.w + bb.w;
      *reinterpret_cast<float4*>(&bufT[row * 128 + c]) = o;
      store_bf4(fragX2 + fragidx(row, c), o);
    }
  }
  __syncthreads();

  // GEMM3: x2(16x128) @ e1_w1(128x256) -> gelu -> fragH2
  for (int c = 0; c < 16; ++c) {
    v8f acc = {};
#pragma unroll
    for (int s = 0; s < 4; ++s)
      acc = wmma_bf16(load_afrag(fragX2, s, lane), load_bfrag(pk_e1w1, c, s, 4, lane), acc);
    int n = c * 16 + dl;
#pragma unroll
    for (int r = 0; r < 8; ++r) {
      int row = r + (hi << 3);
      fragH2[fragidx(row, n)] = (__bf16)gelu_exact(acc[r]);
    }
  }
  __syncthreads();

  // GEMM4: gelu2(16x256) @ e1_w2(256x128) + b2 + residual(x2) -> bufR
  for (int c = 0; c < 8; ++c) {
    v8f acc = {};
#pragma unroll
    for (int s = 0; s < 8; ++s)
      acc = wmma_bf16(load_afrag(fragH2, s, lane), load_bfrag(pk_e1w2, c, s, 8, lane), acc);
    int n = c * 16 + dl;
#pragma unroll
    for (int r = 0; r < 8; ++r) {
      int row = r + (hi << 3);
      bufR[row * 128 + n] = bufT[row * 128 + n] + acc[r] + b2[n];
    }
  }
  __syncthreads();

  // LN1 per row -> global out (float4)
  {
    int row = dl, half = hi;
    float s = 0.f, ss = 0.f;
    for (int c4 = half * 16; c4 < half * 16 + 16; ++c4) {
      float4 v = *reinterpret_cast<const float4*>(&bufR[row * 128 + c4 * 4]);
      s += v.x + v.y + v.z + v.w;
      ss += v.x * v.x + v.y * v.y + v.z * v.z + v.w * v.w;
    }
    s += __shfl_xor(s, 16, 32);
    ss += __shfl_xor(ss, 16, 32);
    float mean = s * (1.f / 128.f);
    float rstd = rsqrtf(ss * (1.f / 128.f) - mean * mean + 1e-5f);
    int j = jt * 16 + row;
    float* orow = out_e + (((size_t)(b * N_ + i)) * N_ + j) * D_;
    for (int c4 = half * 16; c4 < half * 16 + 16; ++c4) {
      int c = c4 * 4;
      float4 v = *reinterpret_cast<const float4*>(&bufR[row * 128 + c]);
      float4 gg = *reinterpret_cast<const float4*>(&sg[256 + c]);
      float4 bb = *reinterpret_cast<const float4*>(&sg[384 + c]);
      float4 o;
      o.x = (v.x - mean) * rstd * gg.x + bb.x;
      o.y = (v.y - mean) * rstd * gg.y + bb.y;
      o.z = (v.z - mean) * rstd * gg.z + bb.z;
      o.w = (v.w - mean) * rstd * gg.w + bb.w;
      *reinterpret_cast<float4*>(orow + c) = o;
    }
  }
}

// ---------- launch ----------
extern "C" void kernel_launch(void* const* d_in, const int* in_sizes, int n_in,
                              void* d_out, int out_size, void* d_ws, size_t ws_size,
                              hipStream_t stream) {
  (void)in_sizes; (void)n_in; (void)out_size; (void)ws_size;
  const float* node    = (const float*)d_in[0];
  const float* edge    = (const float*)d_in[1];
  const float* wqkv_n  = (const float*)d_in[3];
  const float* wqkv_e  = (const float*)d_in[4];
  const float* lin0_w  = (const float*)d_in[5];
  const float* lin0_b  = (const float*)d_in[6];
  const float* ln0_g   = (const float*)d_in[7];
  const float* ln0_bb  = (const float*)d_in[8];
  const float* ln1_g   = (const float*)d_in[9];
  const float* ln1_bb  = (const float*)d_in[10];
  const float* nmlp_w1 = (const float*)d_in[11];
  const float* nmlp_w2 = (const float*)d_in[12];
  const float* nmlp_b2 = (const float*)d_in[13];
  const float* e0_we   = (const float*)d_in[14];
  const float* e0_be   = (const float*)d_in[15];
  const float* e0_ws   = (const float*)d_in[16];
  const float* e0_bs   = (const float*)d_in[17];
  const float* e0_wt   = (const float*)d_in[18];
  const float* e0_bt   = (const float*)d_in[19];
  const float* e0_wer  = (const float*)d_in[20];
  const float* e0_wec  = (const float*)d_in[21];
  const float* e0_w1   = (const float*)d_in[22];
  const float* e0_b1   = (const float*)d_in[23];
  const float* e1_w1   = (const float*)d_in[24];
  const float* e1_w2   = (const float*)d_in[25];
  const float* e1_b2   = (const float*)d_in[26];
  const float* eln0_g  = (const float*)d_in[27];
  const float* eln0_b  = (const float*)d_in[28];
  const float* eln1_g  = (const float*)d_in[29];
  const float* eln1_b  = (const float*)d_in[30];

  char* ws = (char*)d_ws;
  float* qkvn   = (float*)(ws + 0);                 // 1,179,648 B
  float* attn_o = (float*)(ws + 1179648);
  float* mi     = (float*)(ws + 1572864);
  float* mj     = (float*)(ws + 1966080);
  float* sn     = (float*)(ws + 2359296);
  float* tn     = (float*)(ws + 2752512);
  float* se     = (float*)(ws + 3145728);
  float* te     = (float*)(ws + 3538944);
  __bf16* pk_qkv  = (__bf16*)(ws + 3932160);
  __bf16* pk_we   = (__bf16*)(ws + 4063232);
  __bf16* pk_w1   = (__bf16*)(ws + 4128768);
  __bf16* pk_e1w1 = (__bf16*)(ws + 4161536);
  __bf16* pk_e1w2 = (__bf16*)(ws + 4227072);
  float* kn2 = (float*)(ws + 4292608);              // 393,216 B
  float* vn2 = (float*)(ws + 4685824);              // 393,216 B

  float* out_x = (float*)d_out;
  float* out_e = out_x + (size_t)B_ * N_ * D_;

  pack_w<<<(128 * 512 + 255) / 256, 256, 0, stream>>>(wqkv_e, pk_qkv, 128, 512);
  pack_w<<<(256 * 128 + 255) / 256, 256, 0, stream>>>(e0_we, pk_we, 256, 128);
  pack_w<<<(128 * 128 + 255) / 256, 256, 0, stream>>>(e0_w1, pk_w1, 128, 128);
  pack_w<<<(128 * 256 + 255) / 256, 256, 0, stream>>>(e1_w1, pk_e1w1, 128, 256);
  pack_w<<<(256 * 128 + 255) / 256, 256, 0, stream>>>(e1_w2, pk_e1w2, 256, 128);

  node_qkv<<<(B_ * N_ * 3 * D_ + 255) / 256, 256, 0, stream>>>(node, wqkv_n, qkvn);
  qkv_rearr<<<(B_ * H_ * 16 * N_ + 255) / 256, 256, 0, stream>>>(qkvn, kn2, vn2);

  attn_fused<<<B_ * N_, 256, 0, stream>>>(qkvn, kn2, vn2, edge, pk_qkv, attn_o);

  node_update<<<B_ * N_, 128, 0, stream>>>(node, attn_o, lin0_w, lin0_b,
                                           ln0_g, ln0_bb, ln1_g, ln1_bb,
                                           nmlp_w1, nmlp_w2, nmlp_b2, out_x);

  edge_means<<<2 * B_ * N_, 128, 0, stream>>>(edge, mi, mj);

  edge_terms<<<B_ * N_, 128, 0, stream>>>(out_x, mi, mj, e0_ws, e0_bs, e0_wt, e0_bt,
                                          e0_wec, e0_wer, sn, tn, se, te);

  edge_fused<<<B_ * N_ * (N_ / 16), 32, 0, stream>>>(edge, sn, tn, se, te,
                                                     pk_we, e0_be, pk_w1, e0_b1,
                                                     pk_e1w1, pk_e1w2, e1_b2,
                                                     eln0_g, eln0_b, eln1_g, eln1_b, out_e);
}
